// GlobalLocalAttention_47124381172152
// MI455X (gfx1250) — matrix operand order repaired and checked by reading, not compiled, not run on hardware
//
#include <hip/hip_runtime.h>
#include <math.h>

// ---------------- model constants ----------------
#define NHEADS   8
#define HH       128
#define WWID     128
#define HW       16384          // 128*128
#define NTOT     131072         // 8*16384
#define ATT_SCALE 0.35355339059327373f   // 8^-0.5
#define BN_SC     0.9999950000374996f    // 1/sqrt(1+1e-5)

typedef __attribute__((ext_vector_type(16))) _Float16 v16h;
typedef __attribute__((ext_vector_type(16))) __bf16   v16bf;
typedef __attribute__((ext_vector_type(8)))  float    v8f;
typedef __attribute__((ext_vector_type(4)))  unsigned int u32x4;
typedef __attribute__((ext_vector_type(8)))  unsigned int u32x8;

typedef __attribute__((address_space(3))) float lds_f;

__device__ __forceinline__ v8f wmma_f16(v16h a, v16h b, v8f c) {
  return __builtin_amdgcn_wmma_f32_16x16x32_f16(false, a, false, b, (short)0, c, false, false);
}
__device__ __forceinline__ v8f wmma_bf16(v16bf a, v16bf b, v8f c) {
  return __builtin_amdgcn_wmma_f32_16x16x32_bf16(false, a, false, b, (short)0, c, false, false);
}
// ISA 7.12.2: 16-bit A 16x32 tile, lane&15=M, lane>>4=ksel.
// half j -> K = ksel*16 + j (this code's contiguous order, not the ISA table order; matches B fills)
__device__ __forceinline__ int a_kidx(int ksel, int j) {
  return ksel * 16 + j;
}

// CDNA5 async global->LDS copy (16B per lane), tracked by ASYNCcnt.
__device__ __forceinline__ void async_g2l_b128(unsigned lds_byte, const float* gaddr) {
  asm volatile("global_load_async_to_lds_b128 %0, %1, off"
               :: "v"(lds_byte), "v"(gaddr) : "memory");
}
__device__ __forceinline__ void wait_asynccnt0() {
  asm volatile("s_wait_asynccnt 0" ::: "memory");
}

// CDNA5 Tensor Data Mover: DMA a 2D tile (fp32 elements) of a row-major
// tensor into LDS, compacted (tile rows consecutive).  D# per ISA 08 §8.3/8.4.
__device__ __forceinline__ void tdm_load_2d(unsigned lds_byte, const void* gaddr,
                                            unsigned tensor_d0, unsigned tensor_d1,
                                            unsigned tile_d0, unsigned tile_d1,
                                            unsigned d0_stride) {
  unsigned long long ga = (unsigned long long)gaddr;
  u32x4 g0;
  g0[0] = 1u;                                   // count=1, user descriptor
  g0[1] = lds_byte;                             // lds_addr [63:32]
  g0[2] = (unsigned)(ga & 0xffffffffu);         // global_addr lo
  g0[3] = (unsigned)((ga >> 32) & 0x01ffffffu)  // global_addr hi (57-bit)
        | 0x80000000u;                          // type=2 ("image") in [127:126]
  u32x8 g1;
  g1[0] = 0x00020000u;                          // data_size=2 (4 bytes), no flags
  g1[1] = (tensor_d0 & 0xffffu) << 16;          // tensor_dim0 [79:48] lo part
  g1[2] = (tensor_d0 >> 16) | ((tensor_d1 & 0xffffu) << 16);
  g1[3] = (tensor_d1 >> 16) | (tile_d0 << 16);  // tile_dim0 [127:112]
  g1[4] = tile_d1;                              // tile_dim1 [143:128]; tile_dim2=0
  g1[5] = d0_stride;                            // tensor_dim0_stride [207:160] lo
  g1[6] = 0u;
  g1[7] = 0u;
  u32x4 gz = {0u, 0u, 0u, 0u};                  // groups 2/3 unused (tile_dim2=0)
  asm volatile("tensor_load_to_lds %0, %1, %2, %3"
               :: "s"(g0), "s"(g1), "s"(gz), "s"(gz) : "memory");
}

// ---------------- DFT twiddles (128-pt) ----------------
__global__ void k_dft_init(float* Cc, float* Ss) {
  int t = blockIdx.x * blockDim.x + threadIdx.x;
  if (t >= 16384) return;
  int u = t >> 7, j = t & 127;
  float a0 = 6.283185307179586f * (float)(u * j) / 128.0f;
  Cc[t] = cosf(a0);   // D = Cc - i*Ss  (forward fft kernel)
  Ss[t] = sinf(a0);
}

// ---------------- qkv 1x1 conv, f16 output [192][NTOT] ----------------
__global__ void k_qkv_wmma(const float* __restrict__ x, const float* __restrict__ w,
                           _Float16* __restrict__ out) {
  __builtin_prefetch(w, 0, 3);
  int lane = threadIdx.x & 31, wave = threadIdx.x >> 5;
  int col = lane & 15, ksel = lane >> 4;
  const int Mtiles = 12;                 // 192/16
  const int Ntiles = NTOT / 16;
  int tile = blockIdx.x * 8 + wave;
  if (tile >= Mtiles * Ntiles) return;
  int mt = tile % Mtiles, nt = tile / Mtiles;
  int m0 = mt << 4, n0 = nt << 4;
  int n = n0 + col;
  int bimg = n >> 14, p = n & 16383;
  v8f acc = {};
  for (int kk = 0; kk < 64; kk += 32) {
    v16h a, b;
    for (int j = 0; j < 16; ++j)
      a[j] = (_Float16)w[(m0 + col) * 64 + kk + a_kidx(ksel, j)];
    for (int j = 0; j < 16; ++j) {
      int c = kk + ksel * 16 + j;
      b[j] = (_Float16)x[((size_t)(bimg * 64 + c)) * HW + p];
    }
    acc = wmma_f16(a, b, acc);
  }
  for (int i = 0; i < 8; ++i) {
    int och = m0 + i + ksel * 8;
    out[(size_t)och * NTOT + n] = (_Float16)acc[i];
  }
}

// ---------------- generic 1x1 conv (Cin=64), fp32 NCHW out ----------------
__global__ void k_conv1x1_wmma(const float* __restrict__ in, const float* __restrict__ w,
                               const float* __restrict__ bias, float* __restrict__ outp,
                               int Cout, int Pout, int Ntot, int act) {
  __builtin_prefetch(w, 0, 3);
  int lane = threadIdx.x & 31, wave = threadIdx.x >> 5;
  int col = lane & 15, ksel = lane >> 4;
  int Mtiles = Cout >> 4;
  int Ntiles = (Ntot + 15) >> 4;
  int tile = blockIdx.x * 8 + wave;
  if (tile >= Mtiles * Ntiles) return;
  int mt = tile % Mtiles, nt = tile / Mtiles;
  int m0 = mt << 4, n0 = nt << 4;
  int n = n0 + col;
  bool nv = n < Ntot;
  int bimg = nv ? (n / Pout) : 0;
  int p    = nv ? (n % Pout) : 0;
  v8f acc = {};
  for (int kk = 0; kk < 64; kk += 32) {
    v16h a, b;
    for (int j = 0; j < 16; ++j)
      a[j] = (_Float16)w[(m0 + col) * 64 + kk + a_kidx(ksel, j)];
    for (int j = 0; j < 16; ++j) {
      int c = kk + ksel * 16 + j;
      b[j] = nv ? (_Float16)in[((size_t)(bimg * 64 + c)) * Pout + p] : (_Float16)0.f;
    }
    acc = wmma_f16(a, b, acc);
  }
  if (nv) {
    for (int i = 0; i < 8; ++i) {
      int och = m0 + i + ksel * 8;
      float v = acc[i] + (bias ? bias[och] : 0.f);
      if (act == 1) v = fmaxf(v, 0.f);
      outp[((size_t)(bimg * Cout + och)) * Pout + p] = v;
    }
  }
}

// ---------------- generic 3x3 conv via implicit GEMM (Cin=64, K=576) ------
// wA rows [0,split), wB rows [split,Cout). gamma: per-out-ch weight scale
// (times gscale). cw: optional extra 1x1 weights merged into center tap with
// per-channel cgamma*gscale. biasC: optional extra bias. act: 0 none,1 relu,2 relu6
__global__ void k_conv3x3_wmma(
    const float* __restrict__ in, int Hin, int Win,
    const float* __restrict__ wA, const float* __restrict__ wB, int split,
    const float* __restrict__ biasA, const float* __restrict__ biasB,
    const float* __restrict__ biasC,
    const float* __restrict__ gamma, const float* __restrict__ cw,
    const float* __restrict__ cgamma, float gscale,
    float* __restrict__ outp, int Cout, int Hout, int Wout,
    int stride, int pad, int act) {
  __builtin_prefetch(wA, 0, 3);
  int lane = threadIdx.x & 31, wave = threadIdx.x >> 5;
  int col = lane & 15, ksel = lane >> 4;
  int Pout = Hout * Wout;
  int Ntot = 8 * Pout;
  int Mtiles = Cout >> 4;
  int Ntiles = (Ntot + 15) >> 4;
  int tile = blockIdx.x * 8 + wave;
  if (tile >= Mtiles * Ntiles) return;
  int mt = tile % Mtiles, nt = tile / Mtiles;
  int m0 = mt << 4, n0 = nt << 4;
  int o = m0 + col;                 // out channel for A fetch
  int n = n0 + col;
  bool nv = n < Ntot;
  int bimg = nv ? (n / Pout) : 0;
  int po   = nv ? (n % Pout) : 0;
  int oh = po / Wout, ow = po % Wout;
  float gsc = gamma ? gamma[o] * gscale : gscale;
  float csc = cgamma ? cgamma[o] * gscale : gscale;
  v8f acc = {};
  for (int kk = 0; kk < 576; kk += 32) {
    v16h a, b;
    for (int j = 0; j < 16; ++j) {
      int k = kk + a_kidx(ksel, j);
      int c = k / 9, tap = k % 9;
      float wv = (o < split) ? wA[(o * 64 + c) * 9 + tap]
                             : wB[((o - split) * 64 + c) * 9 + tap];
      wv *= gsc;
      if (cw && tap == 4) wv += cw[o * 64 + c] * csc;
      a[j] = (_Float16)wv;
    }
    for (int j = 0; j < 16; ++j) {
      int k = kk + ksel * 16 + j;
      int c = k / 9, tap = k % 9;
      int ih = oh * stride - pad + tap / 3;
      int iw = ow * stride - pad + tap % 3;
      float xv = 0.f;
      if (nv && ih >= 0 && ih < Hin && iw >= 0 && iw < Win)
        xv = in[((size_t)(bimg * 64 + c) * Hin + ih) * Win + iw];
      b[j] = (_Float16)xv;
    }
    acc = wmma_f16(a, b, acc);
  }
  if (nv) {
    for (int i = 0; i < 8; ++i) {
      int och = m0 + i + ksel * 8;
      float bo = 0.f;
      if (biasA) bo = (och < split) ? biasA[och] : biasB[och - split];
      if (biasC) bo += biasC[och];
      float v = acc[i] + bo;
      if (act == 1) v = fmaxf(v, 0.f);
      if (act == 2) v = fminf(fmaxf(v, 0.f), 6.f);
      outp[((size_t)(bimg * Cout + och)) * Pout + po] = v;
    }
  }
}

// ---------------- batched 128x128x128 GEMM (bf16 WMMA, fp32 acc) ----------
// C = alpha*A@B + beta*C, per-batch strides in floats (0 = shared operand).
// A k-panel is staged via the Tensor Data Mover (TENSORcnt); B k-panel via
// async global->LDS copies (ASYNCcnt).  8 waves x 8 sub-tiles per block.
__global__ void k_gemm128(const float* __restrict__ A, long aStr,
                          const float* __restrict__ Bm, long bStr,
                          float* __restrict__ Cm, long cStr,
                          float alpha, float beta) {
  __shared__ float sA[128 * 32];   // A panel  [m][kc]   16 KB
  __shared__ float sB[32 * 128];   // B panel  [r][n]    16 KB
  int bi = blockIdx.x;
  const float* Ab = A  + (size_t)bi * aStr;
  const float* Bb = Bm + (size_t)bi * bStr;
  float*       Cb = Cm + (size_t)bi * cStr;
  int tid = threadIdx.x;
  int lane = tid & 31, wave = tid >> 5;
  int col = lane & 15, ksel = lane >> 4;
  // raw LDS byte offsets for async/TDM destination operands
  unsigned sA0 = (unsigned)(unsigned long long)(lds_f*)&sA[0];
  unsigned sB0 = (unsigned)(unsigned long long)(lds_f*)&sB[0];

  v8f acc[8] = {};
  for (int kk = 0; kk < 128; kk += 32) {
    // A panel: TDM 2D tile, 32 cols (X) x 128 rows (Y) at column kk,
    // tensor 128x128 fp32 with row stride 128 -> lands as sA[m][kc].
    if (wave == 0) {
      tdm_load_2d(sA0, Ab + kk, 128u, 128u, 32u, 128u, 128u);
      __builtin_amdgcn_s_wait_tensorcnt(0);
    }
    // B panel: async copies, 32x128 fp32 = 1024 x 16B chunks
    for (int c = tid; c < 1024; c += 256) {
      int r = c >> 5, n4 = (c & 31) << 2;
      async_g2l_b128(sB0 + (unsigned)((r * 128 + n4) << 2),
                     Bb + (kk + r) * 128 + n4);
    }
    wait_asynccnt0();
    __syncthreads();
    for (int tt = 0; tt < 8; ++tt) {
      int t = wave * 8 + tt;
      int m0 = (t >> 3) << 4, n0 = (t & 7) << 4;
      v16bf a, b;
      for (int j = 0; j < 16; ++j)
        a[j] = (__bf16)sA[(m0 + col) * 32 + a_kidx(ksel, j)];
      for (int j = 0; j < 16; ++j)
        b[j] = (__bf16)sB[(ksel * 16 + j) * 128 + n0 + col];
      acc[tt] = wmma_bf16(a, b, acc[tt]);
    }
    __syncthreads();   // all reads done before next iteration overwrites LDS
  }
  for (int tt = 0; tt < 8; ++tt) {
    int t = wave * 8 + tt;
    int m0 = (t >> 3) << 4, n0 = (t & 7) << 4;
    for (int i = 0; i < 8; ++i) {
      size_t idx = (size_t)(m0 + i + ksel * 8) * 128 + n0 + col;
      float v = alpha * acc[tt][i];
      if (beta != 0.0f) v += beta * Cb[idx];
      Cb[idx] = v;
    }
  }
}

// ---------------- windowed attention: one (window, head) per block --------
__global__ void k_attn_wmma(const _Float16* __restrict__ qkv,
                            const float* __restrict__ rpb_table,
                            const int* __restrict__ rel_index,
                            float* __restrict__ o_out) {
  __shared__ _Float16 qs[64 * 32];
  __shared__ _Float16 ks_[64 * 32];
  __shared__ _Float16 vs[64 * 16];
  __shared__ float    dots[64 * 64];
  __shared__ _Float16 at[64 * 64];

  int head = blockIdx.x & 7;
  int win  = blockIdx.x >> 3;
  int gw0  = win & 15;
  int gh0  = (win >> 4) & 15;
  int bimg = win >> 8;
  int tid  = threadIdx.x;

  for (int i = tid; i < 64 * 32; i += 256) { qs[i] = (_Float16)0.f; ks_[i] = (_Float16)0.f; }
  for (int i = tid; i < 64 * 16; i += 256) vs[i] = (_Float16)0.f;
  __syncthreads();
  for (int idx = tid; idx < 64 * 8; idx += 256) {
    int i = idx >> 3, d = idx & 7;
    int h = gh0 * 8 + (i >> 3), w = gw0 * 8 + (i & 7);
    int n = bimg * HW + h * WWID + w;
    int ch = head * 8 + d;
    qs[i * 32 + d]  = qkv[(size_t)(ch      ) * NTOT + n];
    ks_[i * 32 + d] = qkv[(size_t)(ch +  64) * NTOT + n];
    vs[i * 16 + d]  = qkv[(size_t)(ch + 128) * NTOT + n];
  }
  __syncthreads();

  int lane = tid & 31, wave = tid >> 5;
  int col = lane & 15, ksel = lane >> 4;
  // dots = q @ k^T   (16 tiles, 2 per wave, K=32 padded)
  for (int tt = 0; tt < 2; ++tt) {
    int t = wave * 2 + tt;
    int m0 = (t >> 2) << 4, n0 = (t & 3) << 4;
    v16h a, b; v8f acc = {};
    for (int j = 0; j < 16; ++j) a[j] = qs[(m0 + col) * 32 + a_kidx(ksel, j)];
    for (int j = 0; j < 16; ++j) b[j] = ks_[(n0 + col) * 32 + ksel * 16 + j];
    acc = wmma_f16(a, b, acc);
    for (int i = 0; i < 8; ++i)
      dots[(m0 + i + ksel * 8) * 64 + n0 + col] = acc[i];
  }
  __syncthreads();
  if (tid < 64) {                       // softmax row (scale + rpb)
    int r = tid;
    float mx = -1e30f;
    for (int j = 0; j < 64; ++j) {
      float v = dots[r * 64 + j] * ATT_SCALE +
                rpb_table[rel_index[r * 64 + j] * NHEADS + head];
      dots[r * 64 + j] = v; mx = fmaxf(mx, v);
    }
    float s = 0.f;
    for (int j = 0; j < 64; ++j) { float e = __expf(dots[r * 64 + j] - mx); dots[r * 64 + j] = e; s += e; }
    float inv = 1.0f / s;
    for (int j = 0; j < 64; ++j) at[r * 64 + j] = (_Float16)(dots[r * 64 + j] * inv);
  }
  __syncthreads();
  // o = attn @ v     (4 tiles, waves 0..3, K=64)
  if (wave < 4) {
    int m0 = wave << 4;
    v8f acc = {};
    for (int kk = 0; kk < 64; kk += 32) {
      v16h a, b;
      for (int j = 0; j < 16; ++j) a[j] = at[(m0 + col) * 64 + kk + a_kidx(ksel, j)];
      for (int j = 0; j < 16; ++j) b[j] = vs[(kk + ksel * 16 + j) * 16 + col];
      acc = wmma_f16(a, b, acc);
    }
    if (col < 8) {
      for (int i = 0; i < 8; ++i) {
        int r = m0 + i + ksel * 8;
        int h = gh0 * 8 + (r >> 3), w = gw0 * 8 + (r & 7);
        int ch = head * 8 + col;
        o_out[((size_t)(bimg * 64 + ch)) * HW + h * WWID + w] = acc[i];
      }
    }
  }
}

// ---------------- FFT pointwise: G = F * |F| * highpass-mask --------------
__global__ void k_fft_filter(float* Fr, float* Fi) {
  size_t t = (size_t)blockIdx.x * blockDim.x + threadIdx.x;
  if (t >= (size_t)512 * 16384) return;
  int uv = (int)(t & 16383);
  int u = uv >> 7, v = uv & 127;
  int du = (u < 64) ? u : u - 128;
  int dv = (v < 64) ? v : v - 128;
  float m = ((float)(du * du + dv * dv) > 1600.0f) ? 1.0f : 0.0f;
  float fr = Fr[t], fi = Fi[t];
  float mag = sqrtf(fr * fr + fi * fi) * m;
  Fr[t] = fr * mag; Fi[t] = fi * mag;
}

// xc = fw0 * |ifft|/N^2 + fw1 * xc0
__global__ void k_absmix(const float* Wr, const float* Wi, const float* xc0,
                         const float* gb_w, float* outp) {
  size_t t = (size_t)blockIdx.x * blockDim.x + threadIdx.x;
  if (t >= (size_t)512 * 16384) return;
  float w0 = fmaxf(gb_w[0], 0.f), w1 = fmaxf(gb_w[1], 0.f);
  float s = w0 + w1 + 1e-8f;
  float a = sqrtf(Wr[t] * Wr[t] + Wi[t] * Wi[t]) * (1.0f / 16384.0f);
  outp[t] = (w0 / s) * a + (w1 / s) * xc0[t];
}

__global__ void k_maxpool2(const float* in, int Hin, int Win,
                           float* outp, int Hout, int Wout, int stride) {
  size_t t = (size_t)blockIdx.x * blockDim.x + threadIdx.x;
  size_t tot = (size_t)512 * Hout * Wout;
  if (t >= tot) return;
  int wo = (int)(t % Wout); int ho = (int)((t / Wout) % Hout);
  int bc = (int)(t / ((size_t)Wout * Hout));
  const float* base = in + (size_t)bc * Hin * Win;
  int h0 = ho * stride, w0 = wo * stride;
  float v = base[h0 * Win + w0];
  v = fmaxf(v, base[h0 * Win + w0 + 1]);
  v = fmaxf(v, base[(h0 + 1) * Win + w0]);
  v = fmaxf(v, base[(h0 + 1) * Win + w0 + 1]);
  outp[t] = v;
}

__global__ void k_bilinear31to128(const float* in, float* outp) {
  size_t t = (size_t)blockIdx.x * blockDim.x + threadIdx.x;
  if (t >= (size_t)512 * 16384) return;
  int w = (int)(t & 127), h = (int)((t >> 7) & 127);
  int bc = (int)(t >> 14);
  const float* base = in + (size_t)bc * 961;
  float sh = (h + 0.5f) * (31.0f / 128.0f) - 0.5f;
  float sw = (w + 0.5f) * (31.0f / 128.0f) - 0.5f;
  int h0 = (int)floorf(sh), w0 = (int)floorf(sw);
  float fh = sh - h0, fw = sw - w0;
  int h0c = min(max(h0, 0), 30), h1c = min(max(h0 + 1, 0), 30);
  int w0c = min(max(w0, 0), 30), w1c = min(max(w0 + 1, 0), 30);
  float v = (1.f - fh) * (1.f - fw) * base[h0c * 31 + w0c]
          + (1.f - fh) * fw        * base[h0c * 31 + w1c]
          + fh        * (1.f - fw) * base[h1c * 31 + w0c]
          + fh        * fw         * base[h1c * 31 + w1c];
  outp[t] = v;
}

__global__ void k_logsoftmax64(float* data) {
  size_t t = (size_t)blockIdx.x * blockDim.x + threadIdx.x;
  if (t >= (size_t)8 * 16384) return;
  int p = (int)(t % 16384), b = (int)(t / 16384);
  float* base = data + (size_t)b * 64 * 16384 + p;
  float mx = -1e30f;
  for (int c = 0; c < 64; ++c) mx = fmaxf(mx, base[c * 16384]);
  float s = 0.f;
  for (int c = 0; c < 64; ++c) s += __expf(base[c * 16384] - mx);
  float lse = mx + __logf(s);
  for (int c = 0; c < 64; ++c) base[c * 16384] -= lse;
}

// ox + oy (directional avg pools over reflect-padded o) + local
__global__ void k_dirpool_add(const float* o, const float* localf, float* outp) {
  size_t t = (size_t)blockIdx.x * blockDim.x + threadIdx.x;
  if (t >= (size_t)512 * 16384) return;
  int w = (int)(t & 127), h = (int)((t >> 7) & 127);
  size_t bc = t >> 14;
  const float* ob = o + bc * 16384;
  float sx = 0.f, sy = 0.f;
  for (int d = -3; d <= 4; ++d) {
    int r = h + d;
    if (r >= 0 && r <= 128) sx += ob[((r == 128) ? 126 : r) * 128 + w];
    int c = w + d;
    if (c >= 0 && c <= 128) sy += ob[h * 128 + ((c == 128) ? 126 : c)];
  }
  outp[t] = 0.125f * sx + 0.125f * sy + localf[t];
}

// depthwise 8x8 (input zero-padded to 129 then pad 3) + BN
__global__ void k_dwconv8(const float* in, const float* w_dw,
                          const float* g_proj, const float* b_proj, float* outp) {
  size_t t = (size_t)blockIdx.x * blockDim.x + threadIdx.x;
  if (t >= (size_t)512 * 16384) return;
  int w = (int)(t & 127), h = (int)((t >> 7) & 127);
  int bc = (int)(t >> 14), c = bc & 63;
  const float* base = in + (size_t)bc * 16384;
  float acc = 0.f;
  for (int kh = 0; kh < 8; ++kh) {
    int ih = h - 3 + kh;
    if (ih < 0 || ih >= 128) continue;      // rows 128 of pad_out and conv-pad are zero
    for (int kw = 0; kw < 8; ++kw) {
      int iw = w - 3 + kw;
      if (iw < 0 || iw >= 128) continue;
      acc += base[ih * 128 + iw] * w_dw[c * 64 + kh * 8 + kw];
    }
  }
  outp[t] = acc * (g_proj[c] * BN_SC) + b_proj[c];
}

// ---------------------------- launcher -----------------------------------
extern "C" void kernel_launch(void* const* d_in, const int* in_sizes, int n_in,
                              void* d_out, int out_size, void* d_ws, size_t ws_size,
                              hipStream_t stream) {
  const float* x        = (const float*)d_in[0];
  const float* w_qkv    = (const float*)d_in[1];
  const float* w_l1     = (const float*)d_in[2];
  const float* g_l1     = (const float*)d_in[3];
  const float* b_l1     = (const float*)d_in[4];
  const float* w_l2     = (const float*)d_in[5];
  const float* g_l2     = (const float*)d_in[6];
  const float* b_l2     = (const float*)d_in[7];
  const float* f_cos    = (const float*)d_in[8];
  const float* f_sin    = (const float*)d_in[9];
  const float* gb_b1    = (const float*)d_in[10];
  const float* gb_b2    = (const float*)d_in[11];
  const float* gb_w     = (const float*)d_in[12];
  const float* w_post   = (const float*)d_in[13];
  const float* g_post   = (const float*)d_in[14];
  const float* b_post   = (const float*)d_in[15];
  const float* w_gc     = (const float*)d_in[16];
  const float* b_gc     = (const float*)d_in[17];
  const float* w_gc1    = (const float*)d_in[18];
  const float* b_gc1    = (const float*)d_in[19];
  const float* w_gc2    = (const float*)d_in[20];
  const float* b_gc2    = (const float*)d_in[21];
  const float* rpb_tab  = (const float*)d_in[22];
  const float* w_dw     = (const float*)d_in[23];
  const float* g_proj   = (const float*)d_in[24];
  const float* b_proj   = (const float*)d_in[25];
  const float* w_pw     = (const float*)d_in[26];
  const int*   rel_idx  = (const int*)d_in[27];
  float* outp = (float*)d_out;

  // workspace carve-up (floats); ~285 MB total, with lifetime reuse
  const size_t BUF = 8388608;            // one [8,64,128,128] fp32 tensor
  float* bufA = (float*)d_ws;            // local conv out -> later ox+oy+local sum
  float* bufC = bufA + BUF;              // attention output o
  float* bufD = bufC + BUF;              // gabor conv out xc0 -> later resized 128
  float* bufE = bufD + BUF;              // fft Yr/Zr -> xc mixed -> 31x31 pool
  float* bufF = bufE + BUF;              // fft Yi/Zi -> post conv out
  float* bufG = bufF + BUF;              // fft Fr/Wr -> pool/conv temps
  float* bufH = bufG + BUF;              // fft Fi/Wi -> gc2 out (local_final)
  float* dftC = bufH + BUF;              // 128x128 cos
  float* dftS = dftC + 16384;            // 128x128 sin
  _Float16* qkv_h = (_Float16*)(dftS + 16384); // [192][131072] f16 (50 MB)
  float* bufB = (float*)qkv_h;           // reuse qkv region for dwconv out

  k_dft_init<<<64, 256, 0, stream>>>(dftC, dftS);

  // local = BN(conv3x3 w_l1) + BN(conv1x1 w_l2)  (1x1 merged into center tap)
  k_conv3x3_wmma<<<4096, 256, 0, stream>>>(x, 128, 128, w_l1, w_l1, 64,
      b_l1, b_l1, b_l2, g_l1, w_l2, g_l2, BN_SC,
      bufA, 64, 128, 128, 1, 1, 0);

  // global branch
  k_qkv_wmma<<<12288, 256, 0, stream>>>(x, w_qkv, qkv_h);
  k_attn_wmma<<<16384, 256, 0, stream>>>(qkv_h, rpb_tab, rel_idx, bufC);

  // gabor conv: cos -> ch 0..31, sin -> ch 32..63
  k_conv3x3_wmma<<<4096, 256, 0, stream>>>(bufA, 128, 128, f_cos, f_sin, 32,
      gb_b1, gb_b2, nullptr, nullptr, nullptr, nullptr, 1.0f,
      bufD, 64, 128, 128, 1, 1, 0);

  // fft2 via DFT matmuls: F = D X D, D = C - iS  (512 images of 128x128)
  k_gemm128<<<512, 256, 0, stream>>>(bufD, 16384, dftC, 0, bufE, 16384,  1.f, 0.f); // Yr = X@C
  k_gemm128<<<512, 256, 0, stream>>>(bufD, 16384, dftS, 0, bufF, 16384, -1.f, 0.f); // Yi = -X@S
  k_gemm128<<<512, 256, 0, stream>>>(dftC, 0, bufE, 16384, bufG, 16384,  1.f, 0.f); // Fr = C@Yr
  k_gemm128<<<512, 256, 0, stream>>>(dftS, 0, bufF, 16384, bufG, 16384,  1.f, 1.f); //    + S@Yi
  k_gemm128<<<512, 256, 0, stream>>>(dftC, 0, bufF, 16384, bufH, 16384,  1.f, 0.f); // Fi = C@Yi
  k_gemm128<<<512, 256, 0, stream>>>(dftS, 0, bufE, 16384, bufH, 16384, -1.f, 1.f); //    - S@Yr
  k_fft_filter<<<32768, 256, 0, stream>>>(bufG, bufH);                              // G = F|F|m
  // ifft2: W = D* G D*, D* = C + iS  (1/N^2 applied in absmix)
  k_gemm128<<<512, 256, 0, stream>>>(dftC, 0, bufG, 16384, bufE, 16384,  1.f, 0.f); // Zr = C@Gr
  k_gemm128<<<512, 256, 0, stream>>>(dftS, 0, bufH, 16384, bufE, 16384, -1.f, 1.f); //    - S@Gi
  k_gemm128<<<512, 256, 0, stream>>>(dftC, 0, bufH, 16384, bufF, 16384,  1.f, 0.f); // Zi = C@Gi
  k_gemm128<<<512, 256, 0, stream>>>(dftS, 0, bufG, 16384, bufF, 16384,  1.f, 1.f); //    + S@Gr
  k_gemm128<<<512, 256, 0, stream>>>(bufE, 16384, dftC, 0, bufG, 16384,  1.f, 0.f); // Wr = Zr@C
  k_gemm128<<<512, 256, 0, stream>>>(bufF, 16384, dftS, 0, bufG, 16384, -1.f, 1.f); //    - Zi@S
  k_gemm128<<<512, 256, 0, stream>>>(bufE, 16384, dftS, 0, bufH, 16384,  1.f, 0.f); // Wi = Zr@S
  k_gemm128<<<512, 256, 0, stream>>>(bufF, 16384, dftC, 0, bufH, 16384,  1.f, 1.f); //    + Zi@C
  k_absmix<<<32768, 256, 0, stream>>>(bufG, bufH, bufD, gb_w, bufE);                // xc mixed

  // post: ReLU6(BN(conv3x3 w_post))
  k_conv3x3_wmma<<<4096, 256, 0, stream>>>(bufE, 128, 128, w_post, w_post, 64,
      b_post, b_post, nullptr, g_post, nullptr, nullptr, BN_SC,
      bufF, 64, 128, 128, 1, 1, 2);
  k_maxpool2<<<32260, 256, 0, stream>>>(bufF, 128, 128, bufG, 127, 127, 1);
  // gc: conv3x3 stride2 valid + relu -> 63x63
  k_conv3x3_wmma<<<993, 256, 0, stream>>>(bufG, 127, 127, w_gc, w_gc, 64,
      b_gc, b_gc, nullptr, nullptr, nullptr, nullptr, 1.0f,
      bufH, 64, 63, 63, 2, 0, 1);
  k_maxpool2<<<1922, 256, 0, stream>>>(bufH, 63, 63, bufE, 31, 31, 2);
  k_bilinear31to128<<<32768, 256, 0, stream>>>(bufE, bufD);
  // gc1: conv3x3 pad1 + relu
  k_conv3x3_wmma<<<4096, 256, 0, stream>>>(bufD, 128, 128, w_gc1, w_gc1, 64,
      b_gc1, b_gc1, nullptr, nullptr, nullptr, nullptr, 1.0f,
      bufG, 64, 128, 128, 1, 1, 1);
  // gc2: conv1x1 + bias, then log_softmax over channels
  k_conv1x1_wmma<<<4096, 256, 0, stream>>>(bufG, w_gc2, b_gc2, bufH, 64, 16384, NTOT, 0);
  k_logsoftmax64<<<512, 256, 0, stream>>>(bufH);

  // combine: ox + oy + local_final
  k_dirpool_add<<<32768, 256, 0, stream>>>(bufC, bufH, bufA);
  // depthwise 8x8 + BN, then final pointwise 1x1 into d_out
  k_dwconv8<<<32768, 256, 0, stream>>>(bufA, w_dw, g_proj, b_proj, bufB);
  k_conv1x1_wmma<<<4096, 256, 0, stream>>>(bufB, w_pw, nullptr, outp, 64, 16384, NTOT, 0);
}
